// DecoderRNN_5987184411240
// MI455X (gfx1250) — matrix operand order repaired and checked by reading, not compiled, not run on hardware
//
#include <hip/hip_runtime.h>
#include <math.h>

// Problem constants (from reference)
#define V_   32000
#define E_   512
#define H_   1024
#define B_   32
#define S_   64
#define EOS_ 1
#define K_   1024     // shared inner dim of every GEMM (2E == H == 1024)
#define G3H  3072     // 3*H gate width

typedef __attribute__((ext_vector_type(16))) __bf16 v16bf;
typedef __attribute__((ext_vector_type(8)))  __bf16 v8bf;
typedef __attribute__((ext_vector_type(8)))  float  v8f;

// =====================================================================
// Path A GEMM: bf16 weights + bf16 activations (pre-converted once).
// A (32x1024 bf16, 64KB) is staged into LDS with the CDNA5 async-copy
// path (global_load_async_to_lds_b128 + s_wait_asynccnt), then each wave
// computes one 16-wide N tile x both 16-row M tiles.
// C[32, N] = A[32, K_] @ W[N, K_]^T + bias[N]
// =====================================================================
__global__ void __launch_bounds__(256)
wmma_gemm_bf16(const __bf16* __restrict__ A, const __bf16* __restrict__ Wb,
               const float* __restrict__ bias, float* __restrict__ C, int N)
{
    __shared__ __align__(16) __bf16 shA[B_ * K_];   // 64 KB

    // ---- async stage of A into LDS: 64KB = 256 threads x 16 x b128 ----
    {
        const unsigned tid = threadIdx.x;
        // generic-pointer low 32 bits == LDS byte offset (ISA aperture rule)
        const unsigned ldsbase = (unsigned)(uintptr_t)&shA[0];
        const unsigned long long gbase = (unsigned long long)(uintptr_t)A;
        #pragma unroll
        for (int i = 0; i < 16; ++i) {
            const unsigned off = tid * 16u + (unsigned)i * 4096u;
            const unsigned lds = ldsbase + off;
            const unsigned long long g = gbase + off;
            asm volatile("global_load_async_to_lds_b128 %0, %1, off"
                         :: "v"(lds), "v"(g) : "memory");
        }
        asm volatile("s_wait_asynccnt 0x0" ::: "memory");
        __syncthreads();
    }

    const int lane = threadIdx.x & 31;
    const int wave = threadIdx.x >> 5;
    const int n0   = (blockIdx.x * 8 + wave) * 16;   // exact grids: no tail
    const int hi   = lane >> 4;
    const int l15  = lane & 15;

    // B fragment: lane holds weight row (n0+l15); 16 contiguous bf16 K values
    // per 32-chunk (lanes 0-15: K 0..15, lanes 16-31: K 16..31).
    const __bf16* Wp  = Wb + (size_t)(n0 + l15) * K_ + hi * 16;
    // A fragment (from LDS): two contiguous 8-element K runs per lane.
    const __bf16* a0p = &shA[(unsigned)l15 * K_ + (unsigned)(hi * 16)];
    const __bf16* a1p = a0p + 16 * K_;

    v8f acc0 = {};   // batch rows 0..15
    v8f acc1 = {};   // batch rows 16..31

    #pragma unroll 4
    for (int k0 = 0; k0 < K_; k0 += 32) {
        __builtin_prefetch((const void*)(Wp + k0 + 128), 0, 1);
        v16bf bfr = *(const v16bf*)(Wp + k0);          // 2 x global b128
        v8bf  r00 = *(const v8bf*)(a0p + k0);          // ds b128
        v8bf  r01 = *(const v8bf*)(a0p + k0 + 8);
        v8bf  r10 = *(const v8bf*)(a1p + k0);
        v8bf  r11 = *(const v8bf*)(a1p + k0 + 8);
        v16bf a0, a1;
        #pragma unroll
        for (int i = 0; i < 8; ++i) {
            a0[i] = r00[i]; a0[8 + i] = r01[i];
            a1[i] = r10[i]; a1[8 + i] = r11[i];
        }
        acc0 = __builtin_amdgcn_wmma_f32_16x16x32_bf16(false, a0, false, bfr,
                                                       (short)0, acc0, false, false);
        acc1 = __builtin_amdgcn_wmma_f32_16x16x32_bf16(false, a1, false, bfr,
                                                       (short)0, acc1, false, false);
    }

    const int   col = n0 + l15;
    const float bn  = bias[col];
    #pragma unroll
    for (int r = 0; r < 8; ++r) {
        const int m = r + hi * 8;
        C[(size_t)m        * N + col] = acc0[r] + bn;
        C[(size_t)(m + 16) * N + col] = acc1[r] + bn;
    }
}

// =====================================================================
// Path B GEMM (fallback if workspace can't hold bf16 weights):
// stream fp32 from L2, convert in registers, WMMA bf16.
// =====================================================================
__device__ __forceinline__ v16bf load_frag_contig16(const float* __restrict__ p) {
    v16bf f;
    float4 t0 = *(const float4*)(p + 0);
    float4 t1 = *(const float4*)(p + 4);
    float4 t2 = *(const float4*)(p + 8);
    float4 t3 = *(const float4*)(p + 12);
    f[0]=(__bf16)t0.x; f[1]=(__bf16)t0.y; f[2]=(__bf16)t0.z; f[3]=(__bf16)t0.w;
    f[4]=(__bf16)t1.x; f[5]=(__bf16)t1.y; f[6]=(__bf16)t1.z; f[7]=(__bf16)t1.w;
    f[8]=(__bf16)t2.x; f[9]=(__bf16)t2.y; f[10]=(__bf16)t2.z; f[11]=(__bf16)t2.w;
    f[12]=(__bf16)t3.x; f[13]=(__bf16)t3.y; f[14]=(__bf16)t3.z; f[15]=(__bf16)t3.w;
    return f;
}
__device__ __forceinline__ v16bf load_frag_8x2(const float* __restrict__ p0,
                                               const float* __restrict__ p1) {
    v16bf f;
    float4 t0 = *(const float4*)(p0 + 0);
    float4 t1 = *(const float4*)(p0 + 4);
    float4 t2 = *(const float4*)(p1 + 0);
    float4 t3 = *(const float4*)(p1 + 4);
    f[0]=(__bf16)t0.x; f[1]=(__bf16)t0.y; f[2]=(__bf16)t0.z; f[3]=(__bf16)t0.w;
    f[4]=(__bf16)t1.x; f[5]=(__bf16)t1.y; f[6]=(__bf16)t1.z; f[7]=(__bf16)t1.w;
    f[8]=(__bf16)t2.x; f[9]=(__bf16)t2.y; f[10]=(__bf16)t2.z; f[11]=(__bf16)t2.w;
    f[12]=(__bf16)t3.x; f[13]=(__bf16)t3.y; f[14]=(__bf16)t3.z; f[15]=(__bf16)t3.w;
    return f;
}

__global__ void __launch_bounds__(256)
wmma_gemm_nt(const float* __restrict__ A, const float* __restrict__ W,
             const float* __restrict__ bias, float* __restrict__ C, int N)
{
    const int lane = threadIdx.x & 31;
    const int wave = threadIdx.x >> 5;
    const int n0   = (blockIdx.x * 8 + wave) * 16;
    const int hi   = lane >> 4;
    const int l15  = lane & 15;

    const float* Wp  = W + (size_t)(n0 + l15) * K_ + hi * 16;
    const float* Ap0 = A + (size_t)l15 * K_ + hi * 16;
    const float* Ap1 = Ap0 + (size_t)16 * K_;

    v8f acc0 = {}, acc1 = {};
    #pragma unroll 4
    for (int k0 = 0; k0 < K_; k0 += 32) {
        __builtin_prefetch((const void*)(Wp + k0 + 64), 0, 1);
        v16bf bfr = load_frag_contig16(Wp + k0);
        v16bf a0  = load_frag_8x2(Ap0 + k0, Ap0 + k0 + 8);
        v16bf a1  = load_frag_8x2(Ap1 + k0, Ap1 + k0 + 8);
        acc0 = __builtin_amdgcn_wmma_f32_16x16x32_bf16(false, a0, false, bfr,
                                                       (short)0, acc0, false, false);
        acc1 = __builtin_amdgcn_wmma_f32_16x16x32_bf16(false, a1, false, bfr,
                                                       (short)0, acc1, false, false);
    }
    const int   col = n0 + l15;
    const float bn  = bias[col];
    #pragma unroll
    for (int r = 0; r < 8; ++r) {
        const int m = r + hi * 8;
        C[(size_t)m        * N + col] = acc0[r] + bn;
        C[(size_t)(m + 16) * N + col] = acc1[r] + bn;
    }
}

// ---- one-time fp32 -> bf16 weight conversion (HBM-bandwidth bound) ----
__global__ void cvt_f32_to_bf16(const float* __restrict__ src,
                                __bf16* __restrict__ dst, int n4)
{
    const int i = blockIdx.x * blockDim.x + threadIdx.x;
    if (i < n4) {
        float4 t = ((const float4*)src)[i];
        __bf16* d = dst + (size_t)i * 4;
        d[0] = (__bf16)t.x; d[1] = (__bf16)t.y;
        d[2] = (__bf16)t.z; d[3] = (__bf16)t.w;
    }
}

// x[b,0:512] = emb[kw], x[b,512:1024] = emb[lw]; bf16 copy for path A
__global__ void gather_x(const float* __restrict__ emb,
                         const int* __restrict__ targets,
                         const int* __restrict__ targets_kws,
                         int t, float* __restrict__ x, __bf16* __restrict__ xb)
{
    const int b  = blockIdx.x;
    const int kw = targets_kws[b * S_ + (t + 1)];
    const int lw = targets[b * S_ + t];
    const float* ek = emb + (size_t)kw * E_;
    const float* el = emb + (size_t)lw * E_;
    for (int i = threadIdx.x; i < E_; i += blockDim.x) {
        const float vk = ek[i], vl = el[i];
        x[b * (2 * E_) + i]      = vk;
        x[b * (2 * E_) + E_ + i] = vl;
        if (xb) {
            xb[b * (2 * E_) + i]      = (__bf16)vk;
            xb[b * (2 * E_) + E_ + i] = (__bf16)vl;
        }
    }
}

// PyTorch GRU gate math; gi/gh already include biases (added in GEMM).
__global__ void gru_combine(const float* __restrict__ gi, const float* __restrict__ gh,
                            const float* __restrict__ h,
                            float* __restrict__ hnew, __bf16* __restrict__ hnewb)
{
    const int idx = blockIdx.x * blockDim.x + threadIdx.x;   // B_*H_ = 32768
    const int b = idx >> 10, j = idx & (H_ - 1);
    const float* gib = gi + (size_t)b * G3H;
    const float* ghb = gh + (size_t)b * G3H;
    const float ir = gib[j],          hr = ghb[j];
    const float iz = gib[H_ + j],     hz = ghb[H_ + j];
    const float in = gib[2 * H_ + j], hn = ghb[2 * H_ + j];
    const float r = 1.0f / (1.0f + __expf(-(ir + hr)));
    const float z = 1.0f / (1.0f + __expf(-(iz + hz)));
    const float n = tanhf(in + r * hn);
    const float v = (1.0f - z) * n + z * h[idx];
    hnew[idx] = v;
    if (hnewb) hnewb[idx] = (__bf16)v;
}

// EOS blend into next-step states (+bf16 copies for path A)
__global__ void eos_blend(const float* __restrict__ wh, const float* __restrict__ tmp,
                          const int* __restrict__ targets, int t,
                          float* __restrict__ word_h, float* __restrict__ sent_h,
                          __bf16* __restrict__ word_hb, __bf16* __restrict__ sent_hb)
{
    const int idx = blockIdx.x * blockDim.x + threadIdx.x;
    const int b = idx >> 10;
    const float m  = (targets[b * S_ + (t + 1)] == EOS_) ? 1.0f : 0.0f;
    const float tv = tmp[idx];
    const float ns = sent_h[idx] * (1.0f - m) + tv * m;
    const float nw = wh[idx]     * (1.0f - m) + tv * m;
    sent_h[idx] = ns;
    word_h[idx] = nw;
    if (sent_hb) { sent_hb[idx] = (__bf16)ns; word_hb[idx] = (__bf16)nw; }
}

// One block per batch row: logsumexp over V, gather target log-prob,
// accumulate per-row partial (fixed-order reductions -> deterministic).
__global__ void row_loss(const float* __restrict__ logits,
                         const int* __restrict__ targets,
                         const int* __restrict__ targets_len,
                         int t, float* __restrict__ loss_partial)
{
    const int b = blockIdx.x;
    const float* row = logits + (size_t)b * V_;
    __shared__ float red[256];

    float mx = -INFINITY;
    for (int i = threadIdx.x; i < V_; i += 256) mx = fmaxf(mx, row[i]);
    red[threadIdx.x] = mx; __syncthreads();
    for (int s = 128; s > 0; s >>= 1) {
        if (threadIdx.x < s) red[threadIdx.x] = fmaxf(red[threadIdx.x], red[threadIdx.x + s]);
        __syncthreads();
    }
    mx = red[0]; __syncthreads();

    float sum = 0.0f;
    for (int i = threadIdx.x; i < V_; i += 256) sum += __expf(row[i] - mx);
    red[threadIdx.x] = sum; __syncthreads();
    for (int s = 128; s > 0; s >>= 1) {
        if (threadIdx.x < s) red[threadIdx.x] += red[threadIdx.x + s];
        __syncthreads();
    }
    if (threadIdx.x == 0) {
        const int   nxt   = targets[b * S_ + (t + 1)];
        const float g     = row[nxt] - mx - __logf(red[0]);
        const float valid = (targets_len[b] > (t + 1)) ? 1.0f : 0.0f;
        loss_partial[b] += g * valid;
    }
}

__global__ void init_state(const float* __restrict__ sent_state,
                           float* __restrict__ word_h, float* __restrict__ sent_h,
                           __bf16* __restrict__ word_hb, __bf16* __restrict__ sent_hb,
                           float* __restrict__ loss_partial)
{
    const int idx = blockIdx.x * blockDim.x + threadIdx.x;
    if (idx < B_ * H_) {
        const float v = sent_state[idx];   // NL == 1
        word_h[idx] = v;
        sent_h[idx] = v;
        if (word_hb) { word_hb[idx] = (__bf16)v; sent_hb[idx] = (__bf16)v; }
    }
    if (idx < B_) loss_partial[idx] = 0.0f;
}

__global__ void finalize_loss(const float* __restrict__ loss_partial,
                              float* __restrict__ out)
{
    float s = 0.0f;
    for (int b = 0; b < B_; ++b) s += loss_partial[b];   // fixed order
    out[0] = -s;
}

extern "C" void kernel_launch(void* const* d_in, const int* in_sizes, int n_in,
                              void* d_out, int out_size, void* d_ws, size_t ws_size,
                              hipStream_t stream)
{
    const int*   targets     = (const int*)  d_in[0];
    const int*   targets_kws = (const int*)  d_in[1];
    const float* sent_state  = (const float*)d_in[2];
    const int*   targets_len = (const int*)  d_in[3];
    const float* emb_W = (const float*)d_in[4];
    const float* w_Wih = (const float*)d_in[5];
    const float* w_Whh = (const float*)d_in[6];
    const float* w_bih = (const float*)d_in[7];
    const float* w_bhh = (const float*)d_in[8];
    const float* s_Wih = (const float*)d_in[9];
    const float* s_Whh = (const float*)d_in[10];
    const float* s_bih = (const float*)d_in[11];
    const float* s_bhh = (const float*)d_in[12];
    const float* out_W = (const float*)d_in[13];
    const float* out_b = (const float*)d_in[14];

    // ---- fp32 workspace region (5.54 MB) ----
    float* ws     = (float*)d_ws;
    float* x      = ws;                 // [32, 1024]
    float* wh     = ws + 32768;         // [32, 1024]
    float* tmp    = ws + 65536;         // [32, 1024]
    float* word_h = ws + 98304;         // [32, 1024]
    float* sent_h = ws + 131072;        // [32, 1024]
    float* gi     = ws + 163840;        // [32, 3072]
    float* gh     = ws + 262144;        // [32, 3072]
    float* logits = ws + 360448;        // [32, 32000]
    float* lossp  = ws + 1384448;       // [32]

    // ---- bf16 region (path A): 45,481,984 elems = 90.96 MB ----
    const size_t BF16_BASE_BYTES = 5538816;                 // 256B aligned
    const size_t BF16_ELEMS      = 45481984;
    const size_t PATHA_BYTES     = BF16_BASE_BYTES + BF16_ELEMS * 2;
    const bool   bigws           = (ws_size >= PATHA_BYTES);

    __bf16* bb      = (__bf16*)((char*)d_ws + BF16_BASE_BYTES);
    __bf16* xb      = bigws ? bb + 0        : nullptr;      // [32,1024]
    __bf16* whb     = bigws ? bb + 32768    : nullptr;
    __bf16* word_hb = bigws ? bb + 65536    : nullptr;
    __bf16* sent_hb = bigws ? bb + 98304    : nullptr;
    __bf16* Wih_b   = bb + 131072;                          // [3072,1024]
    __bf16* Whh_b   = bb + 3276800;
    __bf16* sWih_b  = bb + 6422528;
    __bf16* sWhh_b  = bb + 9568256;
    __bf16* outW_b  = bb + 12713984;                        // [32000,1024]

    if (bigws) {
        // one-time weight conversion, ~270 MB of HBM traffic per call
        cvt_f32_to_bf16<<<3072, 256, 0, stream>>>(w_Wih, Wih_b, G3H * K_ / 4);
        cvt_f32_to_bf16<<<3072, 256, 0, stream>>>(w_Whh, Whh_b, G3H * K_ / 4);
        cvt_f32_to_bf16<<<3072, 256, 0, stream>>>(s_Wih, sWih_b, G3H * K_ / 4);
        cvt_f32_to_bf16<<<3072, 256, 0, stream>>>(s_Whh, sWhh_b, G3H * K_ / 4);
        cvt_f32_to_bf16<<<32000, 256, 0, stream>>>(out_W, outW_b, V_ * K_ / 4);
    }

    init_state<<<128, 256, 0, stream>>>(sent_state, word_h, sent_h,
                                        word_hb, sent_hb, lossp);

    for (int t = 0; t < S_ - 1; ++t) {
        gather_x<<<B_, 256, 0, stream>>>(emb_W, targets, targets_kws, t, x, xb);
        if (bigws) {
            wmma_gemm_bf16<<<24, 256, 0, stream>>>(xb,      Wih_b,  w_bih, gi, G3H);
            wmma_gemm_bf16<<<24, 256, 0, stream>>>(word_hb, Whh_b,  w_bhh, gh, G3H);
            gru_combine<<<128, 256, 0, stream>>>(gi, gh, word_h, wh, whb);
            wmma_gemm_bf16<<<250, 256, 0, stream>>>(whb, outW_b, out_b, logits, V_);
            row_loss<<<B_, 256, 0, stream>>>(logits, targets, targets_len, t, lossp);
            wmma_gemm_bf16<<<24, 256, 0, stream>>>(whb,     sWih_b, s_bih, gi, G3H);
            wmma_gemm_bf16<<<24, 256, 0, stream>>>(sent_hb, sWhh_b, s_bhh, gh, G3H);
            gru_combine<<<128, 256, 0, stream>>>(gi, gh, sent_h, tmp, nullptr);
        } else {
            wmma_gemm_nt<<<24, 256, 0, stream>>>(x,      w_Wih, w_bih, gi, G3H);
            wmma_gemm_nt<<<24, 256, 0, stream>>>(word_h, w_Whh, w_bhh, gh, G3H);
            gru_combine<<<128, 256, 0, stream>>>(gi, gh, word_h, wh, nullptr);
            wmma_gemm_nt<<<250, 256, 0, stream>>>(wh, out_W, out_b, logits, V_);
            row_loss<<<B_, 256, 0, stream>>>(logits, targets, targets_len, t, lossp);
            wmma_gemm_nt<<<24, 256, 0, stream>>>(wh,     s_Wih, s_bih, gi, G3H);
            wmma_gemm_nt<<<24, 256, 0, stream>>>(sent_h, s_Whh, s_bhh, gh, G3H);
            gru_combine<<<128, 256, 0, stream>>>(gi, gh, sent_h, tmp, nullptr);
        }
        eos_blend<<<128, 256, 0, stream>>>(wh, tmp, targets, t,
                                           word_h, sent_h, word_hb, sent_hb);
    }

    finalize_loss<<<1, 1, 0, stream>>>(lossp, (float*)d_out);
}
